// TinyLM_44994077392994
// MI455X (gfx1250) — matrix operand order repaired and checked by reading, not compiled, not run on hardware
//
#include <hip/hip_runtime.h>
#include <math.h>

// Model constants (match reference).
#define V_  32000
#define D_  768
#define H_  12
#define L_  4
#define N_  2048
#define DH_ 64

typedef __bf16 bf16;
typedef __bf16 v16bf __attribute__((ext_vector_type(16)));
typedef float  v8f   __attribute__((ext_vector_type(8)));

static __device__ __forceinline__ float gelu_exact(float x) {
  return 0.5f * x * (1.0f + erff(x * 0.70710678118654752f));
}

// Async global->LDS copy of 16 bytes (CDNA5 GLOBAL_LOAD_ASYNC_TO_LDS_B128).
// VDST VGPR = LDS byte address (low 32 bits of flat shared pointer, per the
// ISA aperture rule); VADDR = 64-bit global address.
// Tracked by ASYNCcnt; pair with s_wait_asynccnt before the block barrier.
static __device__ __forceinline__ void async_copy_b128(void* lds_dst, const void* gsrc) {
  unsigned lds_addr = (unsigned)(size_t)lds_dst;
  asm volatile("global_load_async_to_lds_b128 %0, %1, off"
               :: "v"(lds_addr), "v"(gsrc)
               : "memory");
}
static __device__ __forceinline__ void async_wait_all() {
  asm volatile("s_wait_asynccnt 0x0" ::: "memory");
}

// ---------------------------------------------------------------------------
// f32 -> bf16 conversion (weights are supplied f32; WMMA wants bf16).
// ---------------------------------------------------------------------------
__global__ void k_cvt_bf16(const float* __restrict__ src, bf16* __restrict__ dst, int n) {
  for (int i = blockIdx.x * blockDim.x + threadIdx.x; i < n; i += gridDim.x * blockDim.x)
    dst[i] = (bf16)src[i];
}

// ---------------------------------------------------------------------------
// Token embedding gather: x[n,:] = tok[idx[n],:]
// ---------------------------------------------------------------------------
__global__ void k_embed(const int* __restrict__ idx, const float* __restrict__ tok,
                        float* __restrict__ x) {
  int n = blockIdx.x;
  int row = idx[n];
  for (int d = threadIdx.x; d < D_; d += blockDim.x)
    x[(size_t)n * D_ + d] = tok[(size_t)row * D_ + d];
}

// ---------------------------------------------------------------------------
// Row LayerNorm (f32 in) -> bf16 out (next GEMM's A operand).
// One block (256 threads) per token row.
// ---------------------------------------------------------------------------
__global__ __launch_bounds__(256)
void k_ln_bf16(const float* __restrict__ x, const float* __restrict__ w,
               const float* __restrict__ b, bf16* __restrict__ out) {
  __shared__ float red[256];
  const int n = blockIdx.x, tid = threadIdx.x;
  const float* xr = x + (size_t)n * D_;

  float s = 0.f;
  for (int d = tid; d < D_; d += 256) s += xr[d];
  red[tid] = s; __syncthreads();
  for (int st = 128; st > 0; st >>= 1) {
    if (tid < st) red[tid] += red[tid + st];
    __syncthreads();
  }
  float mu = red[0] * (1.0f / D_);
  __syncthreads();

  float vs = 0.f;
  for (int d = tid; d < D_; d += 256) { float t = xr[d] - mu; vs += t * t; }
  red[tid] = vs; __syncthreads();
  for (int st = 128; st > 0; st >>= 1) {
    if (tid < st) red[tid] += red[tid + st];
    __syncthreads();
  }
  float rstd = rsqrtf(red[0] * (1.0f / D_) + 1e-5f);

  for (int d = tid; d < D_; d += 256)
    out[(size_t)n * D_ + d] = (bf16)((xr[d] - mu) * rstd * w[d] + b[d]);
}

// ---------------------------------------------------------------------------
// Tiled bf16 WMMA GEMM: out[M,Nout] = A[M,K] @ W[Nout,K]^T  (both K-major)
// Workgroup = 128 threads = 4 waves; 64x64 output tile; K-step 32 via LDS.
// LDS staging uses CDNA5 async global->LDS copies (ASYNCcnt path), so the
// data never round-trips through VGPRs. Each wave owns a 32x32 subtile
// = 2x2 v_wmma_f32_16x16x32_bf16 accumulators.
// Epilogue flags: bias add, exact GELU, residual add, bf16 vs f32 store.
// ---------------------------------------------------------------------------
template <bool BIAS, bool GELU, bool RESID, bool OUTBF>
__global__ __launch_bounds__(128)
void k_gemm(const bf16* __restrict__ A, const bf16* __restrict__ W,
            const float* __restrict__ bias, const float* __restrict__ resid,
            float* __restrict__ outf, bf16* __restrict__ outb,
            int M, int Nout, int K) {
  constexpr int LS = 40;  // LDS row stride in halves (keeps 16B alignment, skews banks)
  __shared__ __attribute__((aligned(16))) bf16 lA[64 * LS];
  __shared__ __attribute__((aligned(16))) bf16 lB[64 * LS];

  const int nbase = blockIdx.x * 64;
  const int mbase = blockIdx.y * 64;
  const int t = threadIdx.x;
  const int lane = t & 31;
  const int wave = t >> 5;
  const int wm = (wave >> 1) * 32;   // wave's row offset in 64x64 tile
  const int wn = (wave & 1) * 32;    // wave's col offset
  const int lm = lane & 15;          // lane row/col within fragment
  const int lh = lane >> 4;          // lane half-group

  v8f acc[2][2] = {};

  for (int kb = 0; kb < K; kb += 32) {
    // ---- stage 64x32 bf16 tiles of A and W into LDS via async copies ----
    #pragma unroll
    for (int i = 0; i < 2; ++i) {
      int lin = t + i * 128;          // 256 chunks of 8 halves per tile
      int r = lin >> 2;
      int c = (lin & 3) * 8;
      async_copy_b128(&lA[r * LS + c], &A[(size_t)(mbase + r) * K + kb + c]);
      async_copy_b128(&lB[r * LS + c], &W[(size_t)(nbase + r) * K + kb + c]);
    }
    if (kb + 32 < K) {  // hint next K tile into cache (global_prefetch_b8)
      __builtin_prefetch(&A[(size_t)(mbase + (t >> 2)) * K + kb + 32], 0, 1);
      __builtin_prefetch(&W[(size_t)(nbase + (t >> 2)) * K + kb + 32], 0, 1);
    }
    async_wait_all();   // this wave's ASYNCcnt -> 0
    __syncthreads();    // make all waves' LDS writes visible

    // ---- build fragments per ISA layouts ----
    // A 16x32 bf16: lane = row (lm); elements 0..7 -> K {lh*8..+7},
    //               elements 8..15 -> K {16+lh*8..+7}
    v16bf af[2];
    #pragma unroll
    for (int mi = 0; mi < 2; ++mi) {
      const bf16* row = &lA[(wm + mi * 16 + lm) * LS];
      #pragma unroll
      for (int e = 0; e < 8; ++e) {
        af[mi][e]     = row[lh * 8 + e];
        af[mi][e + 8] = row[16 + lh * 8 + e];
      }
    }
    // B 32x16 bf16 (K x N): lane = col (lm); elements 0..7 -> K {lh*8..+7},
    //               elements 8..15 -> K {16+lh*8..+7}
    v16bf wf[2];
    #pragma unroll
    for (int ni = 0; ni < 2; ++ni) {
      const bf16* row = &lB[(wn + ni * 16 + lm) * LS];
      #pragma unroll
      for (int e = 0; e < 16; ++e) wf[ni][e] = row[((e >> 3) << 4) + lh * 8 + (e & 7)];
    }

    #pragma unroll
    for (int mi = 0; mi < 2; ++mi)
      #pragma unroll
      for (int ni = 0; ni < 2; ++ni)
        acc[mi][ni] = __builtin_amdgcn_wmma_f32_16x16x32_bf16(
            false, af[mi], false, wf[ni], (short)0, acc[mi][ni], false, false);

    __syncthreads();
  }

  // ---- epilogue: C layout row = vgpr + 8*lh, col = lm ----
  #pragma unroll
  for (int mi = 0; mi < 2; ++mi)
    #pragma unroll
    for (int ni = 0; ni < 2; ++ni) {
      int col = nbase + wn + ni * 16 + lm;
      #pragma unroll
      for (int r = 0; r < 8; ++r) {
        int row = mbase + wm + mi * 16 + r + 8 * lh;
        float vv = acc[mi][ni][r];
        if (BIAS)  vv += bias[col];
        if (GELU)  vv = gelu_exact(vv);
        if (RESID) vv += resid[(size_t)row * Nout + col];
        if (OUTBF) outb[(size_t)row * Nout + col] = (bf16)vv;
        else       outf[(size_t)row * Nout + col] = vv;
      }
    }
}

// ---------------------------------------------------------------------------
// RoPE + split QKV (f32, token-major [N, 3, H, DH]) into head-major bf16
// q/k/v buffers [H, N, DH]. One block = (n, h), 32 threads = 32 even/odd pairs.
// ---------------------------------------------------------------------------
__global__ __launch_bounds__(32)
void k_rope_split(const float* __restrict__ qkv, bf16* __restrict__ q,
                  bf16* __restrict__ k, bf16* __restrict__ v) {
  const int p = threadIdx.x;          // pair index 0..31
  const int n = blockIdx.x, h = blockIdx.y;
  const float* base = qkv + (size_t)n * 3 * D_;
  const float inv = __powf(10000.0f, -(float)(2 * p) * (1.0f / (float)DH_));
  const float th = (float)n * inv;
  const float c = __cosf(th), s = __sinf(th);
  const int o = h * DH_ + 2 * p;

  float q1 = base[o],          q2 = base[o + 1];
  float k1 = base[D_ + o],     k2 = base[D_ + o + 1];
  float v1 = base[2 * D_ + o], v2 = base[2 * D_ + o + 1];

  size_t dst = ((size_t)h * N_ + n) * DH_ + 2 * p;
  q[dst] = (bf16)(q1 * c - q2 * s); q[dst + 1] = (bf16)(q1 * s + q2 * c);
  k[dst] = (bf16)(k1 * c - k2 * s); k[dst + 1] = (bf16)(k1 * s + k2 * c);
  v[dst] = (bf16)v1;                v[dst + 1] = (bf16)v2;
}

// ---------------------------------------------------------------------------
// Flash attention, one wave per (16-query block, head). Causal.
//   S = Q Kt (bf16 WMMA, 2 K-steps over DH=64), online softmax in C-layout
//   (row stats = 16-lane shuffle reductions), P transposed to A-layout via
//   LDS, O += P V (4 WMMAs covering DH=64). Output bf16 [N, D].
// ---------------------------------------------------------------------------
__global__ __launch_bounds__(32)
void k_flash(const bf16* __restrict__ q, const bf16* __restrict__ k,
             const bf16* __restrict__ v, bf16* __restrict__ out) {
  __shared__ float lp[16 * 34];
  const int qb = blockIdx.x, h = blockIdx.y;
  const int lane = threadIdx.x, lm = lane & 15, lh = lane >> 4;
  const float scale = 0.125f;  // 1/sqrt(64)

  // Q fragments (A layout), DH split 0..31 / 32..63
  v16bf aq[2];
  const bf16* qrow = q + ((size_t)h * N_ + qb * 16 + lm) * DH_;
  #pragma unroll
  for (int f = 0; f < 2; ++f)
    #pragma unroll
    for (int e = 0; e < 8; ++e) {
      aq[f][e]     = qrow[f * 32 + lh * 8 + e];
      aq[f][e + 8] = qrow[f * 32 + 16 + lh * 8 + e];
    }

  float mrun[8], lrun[8];
  v8f oacc[4] = {};
  #pragma unroll
  for (int r = 0; r < 8; ++r) { mrun[r] = -3.0e38f; lrun[r] = 0.f; }

  const int jend = qb * 16 + 15;
  for (int jb = 0; jb <= jend; jb += 32) {
    // ---- S tiles: 16 queries x 32 keys ----
    v8f sacc[2] = {};
    #pragma unroll
    for (int tk = 0; tk < 2; ++tk) {
      const bf16* krow = k + ((size_t)h * N_ + jb + tk * 16 + lm) * DH_;
      #pragma unroll
      for (int d = 0; d < 2; ++d) {
        v16bf bk;
        #pragma unroll
        for (int e = 0; e < 16; ++e) bk[e] = krow[d * 32 + ((e >> 3) << 4) + lh * 8 + (e & 7)];
        sacc[tk] = __builtin_amdgcn_wmma_f32_16x16x32_bf16(
            false, aq[d], false, bk, (short)0, sacc[tk], false, false);
      }
    }

    // ---- scale + causal mask ----
    float sv[2][8];
    #pragma unroll
    for (int tk = 0; tk < 2; ++tk)
      #pragma unroll
      for (int r = 0; r < 8; ++r) {
        int row = qb * 16 + r + 8 * lh;
        int col = jb + tk * 16 + lm;
        float x = sacc[tk][r] * scale;
        sv[tk][r] = (col <= row) ? x : -3.0e38f;
      }

    // ---- online softmax: per-row max/sum (16-lane shuffle reductions) ----
    float sf[8];
    #pragma unroll
    for (int r = 0; r < 8; ++r) {
      float mx = fmaxf(sv[0][r], sv[1][r]);
      #pragma unroll
      for (int msk = 1; msk < 16; msk <<= 1) mx = fmaxf(mx, __shfl_xor(mx, msk, 32));
      float nm = fmaxf(mrun[r], mx);
      sf[r] = __expf(mrun[r] - nm);
      mrun[r] = nm;
      float p0 = __expf(sv[0][r] - nm);
      float p1 = __expf(sv[1][r] - nm);
      sv[0][r] = p0; sv[1][r] = p1;
      float rs = p0 + p1;
      #pragma unroll
      for (int msk = 1; msk < 16; msk <<= 1) rs += __shfl_xor(rs, msk, 32);
      lrun[r] = lrun[r] * sf[r] + rs;
    }
    #pragma unroll
    for (int c = 0; c < 4; ++c)
      #pragma unroll
      for (int r = 0; r < 8; ++r) oacc[c][r] *= sf[r];

    // ---- transpose P (C layout -> A layout) through LDS ----
    #pragma unroll
    for (int tk = 0; tk < 2; ++tk)
      #pragma unroll
      for (int r = 0; r < 8; ++r)
        lp[(r + 8 * lh) * 34 + tk * 16 + lm] = sv[tk][r];
    __syncthreads();
    v16bf pf;
    #pragma unroll
    for (int e = 0; e < 8; ++e) {
      pf[e]     = (bf16)lp[lm * 34 + lh * 8 + e];
      pf[e + 8] = (bf16)lp[lm * 34 + 16 + lh * 8 + e];
    }
    __syncthreads();

    // ---- O += P (16x32) @ V (32x64) ----
    #pragma unroll
    for (int c = 0; c < 4; ++c) {
      v16bf bv;
      #pragma unroll
      for (int e = 0; e < 16; ++e)
        bv[e] = v[((size_t)h * N_ + jb + ((e >> 3) << 4) + lh * 8 + (e & 7)) * DH_ + c * 16 + lm];
      oacc[c] = __builtin_amdgcn_wmma_f32_16x16x32_bf16(
          false, pf, false, bv, (short)0, oacc[c], false, false);
    }
  }

  // ---- normalize and store bf16 [N, D] (token-major for O-proj GEMM) ----
  #pragma unroll
  for (int c = 0; c < 4; ++c)
    #pragma unroll
    for (int r = 0; r < 8; ++r) {
      int row = qb * 16 + r + 8 * lh;
      out[(size_t)row * D_ + h * DH_ + c * 16 + lm] = (bf16)(oacc[c][r] / lrun[r]);
    }
}

// ---------------------------------------------------------------------------
extern "C" void kernel_launch(void* const* d_in, const int* in_sizes, int n_in,
                              void* d_out, int out_size, void* d_ws, size_t ws_size,
                              hipStream_t stream) {
  (void)in_sizes; (void)n_in; (void)out_size; (void)ws_size;

  const int*   idx    = (const int*)  d_in[0];
  const float* tok    = (const float*)d_in[1];
  const float* ln1_w  = (const float*)d_in[2];
  const float* ln1_b  = (const float*)d_in[3];
  const float* qkv_w  = (const float*)d_in[4];
  const float* o_w    = (const float*)d_in[5];
  const float* f1_w   = (const float*)d_in[6];
  const float* f1_b   = (const float*)d_in[7];
  const float* f2_w   = (const float*)d_in[8];
  const float* f2_b   = (const float*)d_in[9];
  const float* ln2_w  = (const float*)d_in[10];
  const float* ln2_b  = (const float*)d_in[11];
  const float* lnf_w  = (const float*)d_in[12];
  const float* lnf_b  = (const float*)d_in[13];
  const float* head_w = (const float*)d_in[14];
  float* logits = (float*)d_out;

  // Workspace carve-up (all rewritten each call; ~160 MB total).
  char* ws = (char*)d_ws;
  size_t off = 0;
  auto take = [&](size_t bytes) -> char* {
    char* p = ws + off;
    off += (bytes + 255) & ~(size_t)255;
    return p;
  };
  float* xf     = (float*)take((size_t)N_ * D_ * 4);
  bf16*  hbf    = (bf16*) take((size_t)N_ * D_ * 2);
  float* qkvf   = (float*)take((size_t)N_ * 3 * D_ * 4);
  bf16*  qbf    = (bf16*) take((size_t)N_ * D_ * 2);
  bf16*  kbf    = (bf16*) take((size_t)N_ * D_ * 2);
  bf16*  vbf    = (bf16*) take((size_t)N_ * D_ * 2);
  bf16*  attnbf = (bf16*) take((size_t)N_ * D_ * 2);
  bf16*  ffbf   = (bf16*) take((size_t)N_ * 4 * D_ * 2);
  bf16*  qkvw   = (bf16*) take((size_t)L_ * 3 * D_ * D_ * 2);
  bf16*  ow     = (bf16*) take((size_t)L_ * D_ * D_ * 2);
  bf16*  f1w    = (bf16*) take((size_t)L_ * 4 * D_ * D_ * 2);
  bf16*  f2w    = (bf16*) take((size_t)L_ * D_ * 4 * D_ * 2);
  bf16*  hw     = (bf16*) take((size_t)V_ * D_ * 2);

  // Convert weights f32 -> bf16 (deterministic each call).
  k_cvt_bf16<<<2048, 256, 0, stream>>>(qkv_w,  qkvw, L_ * 3 * D_ * D_);
  k_cvt_bf16<<<2048, 256, 0, stream>>>(o_w,    ow,   L_ * D_ * D_);
  k_cvt_bf16<<<2048, 256, 0, stream>>>(f1_w,   f1w,  L_ * 4 * D_ * D_);
  k_cvt_bf16<<<2048, 256, 0, stream>>>(f2_w,   f2w,  L_ * D_ * 4 * D_);
  k_cvt_bf16<<<2048, 256, 0, stream>>>(head_w, hw,   V_ * D_);

  k_embed<<<N_, 256, 0, stream>>>(idx, tok, xf);

  for (int l = 0; l < L_; ++l) {
    // LN1 -> QKV GEMM (f32 out)
    k_ln_bf16<<<N_, 256, 0, stream>>>(xf, ln1_w + l * D_, ln1_b + l * D_, hbf);
    k_gemm<false, false, false, false><<<dim3(3 * D_ / 64, N_ / 64), 128, 0, stream>>>(
        hbf, qkvw + (size_t)l * 3 * D_ * D_, nullptr, nullptr, qkvf, nullptr,
        N_, 3 * D_, D_);

    // RoPE + split, flash attention
    k_rope_split<<<dim3(N_, H_), 32, 0, stream>>>(qkvf, qbf, kbf, vbf);
    k_flash<<<dim3(N_ / 16, H_), 32, 0, stream>>>(qbf, kbf, vbf, attnbf);

    // O projection with residual: x = x + attn @ o_w^T
    k_gemm<false, false, true, false><<<dim3(D_ / 64, N_ / 64), 128, 0, stream>>>(
        attnbf, ow + (size_t)l * D_ * D_, nullptr, xf, xf, nullptr, N_, D_, D_);

    // LN2 -> FFN1 (bias + GELU, bf16 out) -> FFN2 (bias + residual)
    k_ln_bf16<<<N_, 256, 0, stream>>>(xf, ln2_w + l * D_, ln2_b + l * D_, hbf);
    k_gemm<true, true, false, true><<<dim3(4 * D_ / 64, N_ / 64), 128, 0, stream>>>(
        hbf, f1w + (size_t)l * 4 * D_ * D_, f1_b + l * 4 * D_, nullptr, nullptr, ffbf,
        N_, 4 * D_, D_);
    k_gemm<true, false, true, false><<<dim3(D_ / 64, N_ / 64), 128, 0, stream>>>(
        ffbf, f2w + (size_t)l * D_ * 4 * D_, f2_b + l * D_, xf, xf, nullptr,
        N_, D_, 4 * D_);
  }

  // Final LN -> head GEMM into d_out (2048 x 32000 f32)
  k_ln_bf16<<<N_, 256, 0, stream>>>(xf, lnf_w, lnf_b, hbf);
  k_gemm<false, false, false, false><<<dim3(V_ / 64, N_ / 64), 128, 0, stream>>>(
      hbf, hw, nullptr, nullptr, logits, nullptr, N_, V_, D_);
}
